// Pooling_23974507446587
// MI455X (gfx1250) — matrix operand order repaired and checked by reading, not compiled, not run on hardware
//
#include <hip/hip_runtime.h>

// ---------------------------------------------------------------------------
// Types for CDNA5 WMMA / TDM
// ---------------------------------------------------------------------------
typedef __attribute__((ext_vector_type(16))) __bf16 v16bf;
typedef __attribute__((ext_vector_type(8)))  float  v8f;
typedef __attribute__((ext_vector_type(4)))  unsigned tdm_v4u;
typedef __attribute__((ext_vector_type(8)))  int      tdm_v8i;
typedef __attribute__((ext_vector_type(4)))  int      tdm_v4i;

#if __has_builtin(__builtin_amdgcn_tensor_load_to_lds)
#define HAVE_TDM 1
#endif

// ---------------------------------------------------------------------------
// bf16 <-> f32 helpers (bit manipulation; RNE)
// ---------------------------------------------------------------------------
__device__ inline __bf16 f2bf(float f) {
  unsigned u = __float_as_uint(f);
  u = (u + 0x7FFFu + ((u >> 16) & 1u)) >> 16;
  unsigned short s = (unsigned short)u;
  __bf16 r;
  __builtin_memcpy(&r, &s, 2);
  return r;
}
__device__ inline float bf2f(__bf16 h) {
  unsigned short s;
  __builtin_memcpy(&s, &h, 2);
  return __uint_as_float(((unsigned)s) << 16);
}

__device__ inline float gelu_f(float x) {
  // tanh approximation (jax.nn.gelu default approximate=True)
  float x3 = x * x * x;
  return 0.5f * x * (1.0f + tanhf(0.7978845608028654f * (x + 0.044715f * x3)));
}

// ---------------------------------------------------------------------------
// Cast f32 -> bf16
// ---------------------------------------------------------------------------
__global__ void cast_f32_bf16(const float* __restrict__ src, __bf16* __restrict__ dst, long n) {
  long i = (long)blockIdx.x * blockDim.x + threadIdx.x;
  if (i < n) dst[i] = f2bf(src[i]);
}

// ---------------------------------------------------------------------------
// Fill constant
// ---------------------------------------------------------------------------
__global__ void fill_f32(float* __restrict__ p, float v, long n) {
  long i = (long)blockIdx.x * blockDim.x + threadIdx.x;
  if (i < n) p[i] = v;
}

// ---------------------------------------------------------------------------
// GEMM + bias + GELU, bf16 in / bf16 out, f32 accumulate via WMMA.
// Block tile 128x128, K-step 32, 256 threads = 8 wave32 arranged 2(M)x4(N),
// each wave does 4x2 = 8 v_wmma_f32_16x16x32_bf16 per K-step.
// A tile staged by the Tensor Data Mover (tensor_load_to_lds, TENSORcnt);
// B tile staged transposed so B fragments are contiguous ds_load_b128 pairs.
// ---------------------------------------------------------------------------
#define TM 128
#define TN 128
#define TK 32

__global__ void __launch_bounds__(256)
gemm_bf16_gelu(const __bf16* __restrict__ A, const __bf16* __restrict__ W,
               const float* __restrict__ bias, __bf16* __restrict__ out,
               int M, int N, int K)
{
  __shared__ __bf16 As[TM * TK];   // [m][k]               8 KB
  __shared__ __bf16 Bs[TN * TK];   // [n][k] (transposed)  8 KB

  const int tid  = threadIdx.x;
  const int lane = tid & 31;
  const int wave = tid >> 5;
  const int m0 = blockIdx.x * TM;
  const int n0 = blockIdx.y * TN;
  const int wm = (wave >> 2) * 64;  // 0 / 64
  const int wn = (wave & 3) * 32;   // 0 / 32 / 64 / 96

  v8f zero = {};
  v8f acc[4][2];
#pragma unroll
  for (int mi = 0; mi < 4; mi++)
#pragma unroll
    for (int ni = 0; ni < 2; ni++) acc[mi][ni] = zero;

  // staging indices
  const int bk   = tid >> 3;        // 0..31
  const int bns  = (tid & 7) * 16;  // 0..112
#ifndef HAVE_TDM
  const int arow = tid >> 1;        // 0..127
  const int akh  = (tid & 1) * 16;  // 0 / 16
#endif

  const int r16 = lane & 15;
  const int hi  = lane >> 4;

  for (int kc = 0; kc < K; kc += TK) {
    // ---- stage A tile: 128 rows x 32 K of bf16 ----
#ifdef HAVE_TDM
    if (wave == 0) {
      // Tensor DMA descriptor (D#), ISA 8.3/8.4. 2-D tensor, groups 2/3 zero.
      unsigned ldsA = (unsigned)(size_t)(&As[0]);
      unsigned long long ga = (unsigned long long)(size_t)(A + (size_t)m0 * K + kc);
      tdm_v4u g0 = {
        1u,                                   // count=1, is_restore=0, gather off
        ldsA,                                 // lds_addr (bytes)
        (unsigned)(ga & 0xFFFFFFFFu),         // global_addr[31:0]
        (unsigned)(((ga >> 32) & 0x01FFFFFFull) | 0x80000000u) // addr[56:32] | type=2
      };
      unsigned td1 = (unsigned)(M - m0);      // rows >= M read back zero (OOB)
      unsigned long long st0 = (unsigned long long)K;  // dim0 stride (elements)
      tdm_v8i g1 = {
        (int)(1u << 16),                                        // data_size=1 (2B), wg_mask=0
        (int)((32u & 0xFFFFu) << 16),                           // tensor_dim0[15:0]=32
        (int)(((32u >> 16) & 0xFFFFu) | ((td1 & 0xFFFFu) << 16)),// td0[31:16] | td1[15:0]
        (int)(((td1 >> 16) & 0xFFFFu) | (32u << 16)),           // td1[31:16] | tile_dim0=32
        (int)(128u),                                            // tile_dim1=128, tile_dim2=0
        (int)(st0 & 0xFFFFFFFFu),                               // stride0[31:0]
        (int)((st0 >> 32) & 0xFFFFu),                           // stride0[47:32] | stride1[15:0]
        0                                                       // stride1[47:16]
      };
      tdm_v4i gz4 = {0, 0, 0, 0};
      tdm_v8i gz8 = {0, 0, 0, 0, 0, 0, 0, 0};
      // 6-arg form (clang-23 / therock-10.0 headers)
      __builtin_amdgcn_tensor_load_to_lds(g0, g1, gz4, gz4, gz8, 0);
      __builtin_amdgcn_s_wait_tensorcnt((short)0);
    }
#else
    {
      int grow = m0 + arow;
      uint4 a0 = {0u,0u,0u,0u}, a1 = {0u,0u,0u,0u};
      if (grow < M) {
        const uint4* p = (const uint4*)(A + (size_t)grow * K + kc + akh);
        a0 = p[0];
        a1 = p[1];
      }
      uint4* s = (uint4*)(As + arow * TK + akh);
      s[0] = a0; s[1] = a1;
    }
#endif
    // ---- stage B tile transposed: Bs[n][k] ----
    {
      const uint4* p = (const uint4*)(W + (size_t)(kc + bk) * N + n0 + bns);
      union { uint4 u[2]; __bf16 h[16]; } b;
      b.u[0] = p[0]; b.u[1] = p[1];
      if (kc + TK < K)  // prefetch next K-chunk (global_prefetch_b8)
        __builtin_prefetch(W + (size_t)(kc + TK + bk) * N + n0 + bns, 0, 1);
#pragma unroll
      for (int i = 0; i < 16; i++) Bs[(bns + i) * TK + bk] = b.h[i];
    }
    __syncthreads();

    // ---- fragments per ISA 7.12.2 layouts ----
    v16bf afrag[4], bfrag[2];
#pragma unroll
    for (int mi = 0; mi < 4; mi++) {
      int r  = wm + mi * 16 + r16;
      int kb = hi * 8;  // lanes 0-15: K 0-7 & 16-23 ; lanes 16-31: K 8-15 & 24-31
      union { uint4 u[2]; v16bf v; } f;
      f.u[0] = *(const uint4*)(As + r * TK + kb);
      f.u[1] = *(const uint4*)(As + r * TK + kb + 16);
      afrag[mi] = f.v;
    }
#pragma unroll
    for (int ni = 0; ni < 2; ni++) {
      int c  = wn + ni * 16 + r16;
      int kb = hi * 8;  // lanes 0-15: K 0-7 & 16-23 ; lanes 16-31: K 8-15 & 24-31 (one column)
      union { uint4 u[2]; v16bf v; } f;
      f.u[0] = *(const uint4*)(Bs + c * TK + kb);
      f.u[1] = *(const uint4*)(Bs + c * TK + kb + 16);
      bfrag[ni] = f.v;
    }

#pragma unroll
    for (int mi = 0; mi < 4; mi++)
#pragma unroll
      for (int ni = 0; ni < 2; ni++)
        acc[mi][ni] = __builtin_amdgcn_wmma_f32_16x16x32_bf16(
            false, afrag[mi], false, bfrag[ni], (short)0, acc[mi][ni], false, false);
    __syncthreads();
  }

  // ---- epilogue: bias + GELU -> bf16 ----
#pragma unroll
  for (int mi = 0; mi < 4; mi++)
#pragma unroll
    for (int ni = 0; ni < 2; ni++)
#pragma unroll
      for (int i = 0; i < 8; i++) {
        int row = m0 + wm + mi * 16 + i + hi * 8;
        int col = n0 + wn + ni * 16 + r16;
        if (row < M) {
          float x = acc[mi][ni][i] + bias[col];
          out[(size_t)row * N + col] = f2bf(gelu_f(x));
        }
      }
}

// ---------------------------------------------------------------------------
// Final layer: emb[n, j] = sum_k H[n,k] * W4[k,j] + b4[j]   (K=1024, 24 outs)
// One wave per node; lanes split K, butterfly reduce.
// ---------------------------------------------------------------------------
__global__ void __launch_bounds__(128)
final_layer(const __bf16* __restrict__ H, const float* __restrict__ W4,
            const float* __restrict__ b4, float* __restrict__ emb, int M)
{
  int wave = threadIdx.x >> 5;
  int lane = threadIdx.x & 31;
  int n = blockIdx.x * 4 + wave;
  if (n >= M) return;
  float p[24];
#pragma unroll
  for (int j = 0; j < 24; j++) p[j] = 0.f;
  for (int k = lane; k < 1024; k += 32) {
    float h = bf2f(H[(size_t)n * 1024 + k]);
    const float* w = W4 + (size_t)k * 24;
#pragma unroll
    for (int j = 0; j < 24; j++) p[j] += h * w[j];
  }
#pragma unroll
  for (int j = 0; j < 24; j++)
#pragma unroll
    for (int off = 16; off > 0; off >>= 1)
      p[j] += __shfl_xor(p[j], off, 32);
  if (lane < 24) emb[(size_t)n * 24 + lane] = p[lane] + b4[lane];
}

// ---------------------------------------------------------------------------
// scatter-mean of emb into 1024 super nodes
// ---------------------------------------------------------------------------
__global__ void scatter_add_k(const float* __restrict__ emb, const int* __restrict__ cluster,
                              float* __restrict__ ssum, float* __restrict__ cnt, int n)
{
  int i = blockIdx.x * blockDim.x + threadIdx.x;
  if (i >= n) return;
  int c0 = cluster[i];       // row 0: node index
  int c1 = cluster[n + i];   // row 1: cluster index
#pragma unroll
  for (int j = 0; j < 24; j++)
    atomicAdd(&ssum[c1 * 24 + j], emb[(size_t)c0 * 24 + j]);
  atomicAdd(&cnt[c1], 1.0f);
}

__global__ void finish_mean_k(float* __restrict__ semb, const float* __restrict__ cnt, int total)
{
  int i = blockIdx.x * blockDim.x + threadIdx.x;
  if (i >= total) return;
  float c = cnt[i / 24];
  if (c < 1.f) c = 1.f;
  semb[i] /= c;
}

// ---------------------------------------------------------------------------
// kNN node->cluster (k=5). Cluster table (1024x24 f32 = 96KB) cached in LDS.
// ---------------------------------------------------------------------------
__global__ void __launch_bounds__(128)
knn_node_cluster(const float* __restrict__ emb, const float* __restrict__ semb,
                 int* __restrict__ bidx, int M)
{
  extern __shared__ float sc[]; // 1024*24
  for (int i = threadIdx.x; i < 1024 * 24; i += blockDim.x) sc[i] = semb[i];
  __syncthreads();
  int n = blockIdx.x * blockDim.x + threadIdx.x;
  if (n >= M) return;
  float e[24];
#pragma unroll
  for (int j = 0; j < 24; j++) e[j] = emb[(size_t)n * 24 + j];
  float bd[5]; int bi[5];
#pragma unroll
  for (int t = 0; t < 5; t++) { bd[t] = 3.4e38f; bi[t] = 0; }
  for (int c = 0; c < 1024; c++) {
    const float* s = sc + c * 24;
    float d = 0.f;
#pragma unroll
    for (int j = 0; j < 24; j++) { float t = e[j] - s[j]; d += t * t; }
    if (d < bd[4]) {
      int p = 4;
      while (p > 0 && d < bd[p - 1]) { bd[p] = bd[p - 1]; bi[p] = bi[p - 1]; p--; }
      bd[p] = d; bi[p] = c;
    }
  }
#pragma unroll
  for (int t = 0; t < 5; t++) bidx[n * 5 + t] = bi[t];
}

// ---------------------------------------------------------------------------
// kNN cluster->cluster (k=10, self included)
// ---------------------------------------------------------------------------
__global__ void __launch_bounds__(128)
knn_cluster_cluster(const float* __restrict__ semb, int* __restrict__ sidx)
{
  extern __shared__ float sc[]; // 1024*24
  for (int i = threadIdx.x; i < 1024 * 24; i += blockDim.x) sc[i] = semb[i];
  __syncthreads();
  int n = blockIdx.x * blockDim.x + threadIdx.x;
  if (n >= 1024) return;
  const float* e = sc + n * 24;
  float bd[10]; int bi[10];
#pragma unroll
  for (int t = 0; t < 10; t++) { bd[t] = 3.4e38f; bi[t] = 0; }
  for (int c = 0; c < 1024; c++) {
    const float* s = sc + c * 24;
    float d = 0.f;
#pragma unroll
    for (int j = 0; j < 24; j++) { float t = e[j] - s[j]; d += t * t; }
    if (d < bd[9]) {
      int p = 9;
      while (p > 0 && d < bd[p - 1]) { bd[p] = bd[p - 1]; bi[p] = bi[p - 1]; p--; }
      bd[p] = d; bi[p] = c;
    }
  }
#pragma unroll
  for (int t = 0; t < 10; t++) sidx[n * 10 + t] = bi[t];
}

// ---------------------------------------------------------------------------
// bgraph: assemble edges (kNN + original cluster edges), logits, exp-weights,
// per-src denominator, then normalize.
// ---------------------------------------------------------------------------
__global__ void bgraph_pass1(const float* __restrict__ emb, const float* __restrict__ semb,
                             const int* __restrict__ bidx, const int* __restrict__ cluster,
                             float* __restrict__ graph_o, float* __restrict__ logits_o,
                             float* __restrict__ w_o, float* __restrict__ denom,
                             int nE, int M)
{
  int e = blockIdx.x * blockDim.x + threadIdx.x;
  if (e >= nE) return;
  int src, dst;
  if (e < M * 5) { src = e / 5; dst = bidx[e]; }
  else { int i = e - M * 5; src = cluster[i]; dst = cluster[M + i]; }
  float d = 0.f;
#pragma unroll
  for (int j = 0; j < 24; j++) {
    float t = emb[(size_t)src * 24 + j] - semb[dst * 24 + j];
    d += t * t;
  }
  if (d < 1e-12f) d = 1e-12f;
  float lg = -logf(d);
  float wt = expf(lg);
  graph_o[e] = (float)src;
  graph_o[(size_t)nE + e] = (float)dst;
  logits_o[e] = lg;
  w_o[e] = wt;
  atomicAdd(&denom[src], wt);
}

__global__ void norm_pass2(const float* __restrict__ graph_o, float* __restrict__ w_o,
                           const float* __restrict__ denom, int nE)
{
  int e = blockIdx.x * blockDim.x + threadIdx.x;
  if (e >= nE) return;
  int src = (int)graph_o[e];
  w_o[e] = w_o[e] / (denom[src] + 1e-12f);
}

// ---------------------------------------------------------------------------
// sgraph: symmetrized cluster kNN, sigmoid weights, normalize over src.
// ---------------------------------------------------------------------------
__global__ void sgraph_pass1(const float* __restrict__ semb, const int* __restrict__ sidx,
                             float* __restrict__ graph_o, float* __restrict__ w_o,
                             float* __restrict__ denom, int nHalf)
{
  int e = blockIdx.x * blockDim.x + threadIdx.x;
  int nE = 2 * nHalf;
  if (e >= nE) return;
  int src, dst;
  if (e < nHalf) { src = e / 10; dst = sidx[e]; }
  else { int i = e - nHalf; src = sidx[i]; dst = i / 10; }
  float d = 0.f;
#pragma unroll
  for (int j = 0; j < 24; j++) {
    float t = semb[src * 24 + j] - semb[dst * 24 + j];
    d += t * t;
  }
  if (d < 1e-12f) d = 1e-12f;
  float lg = -logf(d);
  float wt = 1.0f / (1.0f + expf(-lg));  // sigmoid
  graph_o[e] = (float)src;
  graph_o[(size_t)nE + e] = (float)dst;
  w_o[e] = wt;
  atomicAdd(&denom[src], wt);
}

// ---------------------------------------------------------------------------
// host-side launch
// ---------------------------------------------------------------------------
extern "C" void kernel_launch(void* const* d_in, const int* in_sizes, int n_in,
                              void* d_out, int out_size, void* d_ws, size_t ws_size,
                              hipStream_t stream)
{
  const float* nodes = (const float*)d_in[0];
  const float* W1 = (const float*)d_in[1];
  const float* b1 = (const float*)d_in[2];
  const float* W2 = (const float*)d_in[3];
  const float* b2 = (const float*)d_in[4];
  const float* W3 = (const float*)d_in[5];
  const float* b3 = (const float*)d_in[6];
  const float* W4 = (const float*)d_in[7];
  const float* b4 = (const float*)d_in[8];
  const int* graph   = (const int*)d_in[9];  (void)graph;
  const int* cluster = (const int*)d_in[10];

  const int N   = in_sizes[10] / 2;    // 50000 nodes
  const int D   = 256;
  const int HD  = 1024;
  const int EMBD = 24;
  const int C   = 1024;
  const int E   = in_sizes[9] / 2;     // 300000 input edges (mask length)
  const int nBE = N * 5 + N;           // kNN edges + original cluster edges
  const int nSE = 2 * C * 10;          // symmetrized cluster kNN edges

  // ---- workspace carve-out ----
  char* wsb = (char*)d_ws;
  size_t off = 0;
  auto take = [&](size_t bytes) -> char* {
    char* p = wsb + off;
    off = (off + bytes + 255) & ~(size_t)255;
    return p;
  };
  __bf16* Xbf  = (__bf16*)take((size_t)N * D * 2);
  __bf16* W1bf = (__bf16*)take((size_t)D * HD * 2);
  __bf16* W2bf = (__bf16*)take((size_t)HD * HD * 2);
  __bf16* W3bf = (__bf16*)take((size_t)HD * HD * 2);
  __bf16* H0   = (__bf16*)take((size_t)N * HD * 2);
  __bf16* H1   = (__bf16*)take((size_t)N * HD * 2);
  float*  cnt    = (float*)take((size_t)C * 4);
  int*    bidx   = (int*)take((size_t)N * 5 * 4);
  int*    sidx   = (int*)take((size_t)C * 10 * 4);
  float*  denomB = (float*)take((size_t)N * 4);
  float*  denomS = (float*)take((size_t)C * 4);

  // ---- output layout (flat, return order) ----
  float* out      = (float*)d_out;
  float* emb_o    = out;                              // N*24
  float* semb_o   = emb_o    + (size_t)N * EMBD;      // C*24
  float* bgraph_o = semb_o   + (size_t)C * EMBD;      // 2*nBE
  float* bw_o     = bgraph_o + (size_t)2 * nBE;       // nBE
  float* sgraph_o = bw_o     + (size_t)nBE;           // 2*nSE
  float* sw_o     = sgraph_o + (size_t)2 * nSE;       // nSE
  float* logits_o = sw_o     + (size_t)nSE;           // nBE
  float* mask_o   = logits_o + (size_t)nBE;           // E

  auto cdiv = [](long a, long b) { return (int)((a + b - 1) / b); };

  // ---- casts to bf16 ----
  cast_f32_bf16<<<cdiv((long)N * D, 256), 256, 0, stream>>>(nodes, Xbf, (long)N * D);
  cast_f32_bf16<<<cdiv((long)D * HD, 256), 256, 0, stream>>>(W1, W1bf, (long)D * HD);
  cast_f32_bf16<<<cdiv((long)HD * HD, 256), 256, 0, stream>>>(W2, W2bf, (long)HD * HD);
  cast_f32_bf16<<<cdiv((long)HD * HD, 256), 256, 0, stream>>>(W3, W3bf, (long)HD * HD);

  // ---- MLP: three WMMA GEMM+GELU layers, then f32 output layer ----
  dim3 gemm_grid(cdiv(N, TM), HD / TN);
  gemm_bf16_gelu<<<gemm_grid, 256, 0, stream>>>(Xbf, W1bf, b1, H0, N, HD, D);
  gemm_bf16_gelu<<<gemm_grid, 256, 0, stream>>>(H0,  W2bf, b2, H1, N, HD, HD);
  gemm_bf16_gelu<<<gemm_grid, 256, 0, stream>>>(H1,  W3bf, b3, H0, N, HD, HD);
  final_layer<<<cdiv(N, 4), 128, 0, stream>>>(H0, W4, b4, emb_o, N);

  // ---- scatter mean -> semb ----
  fill_f32<<<cdiv((long)C * EMBD, 256), 256, 0, stream>>>(semb_o, 0.f, (long)C * EMBD);
  fill_f32<<<cdiv((long)C, 256), 256, 0, stream>>>(cnt, 0.f, (long)C);
  scatter_add_k<<<cdiv(N, 256), 256, 0, stream>>>(emb_o, cluster, semb_o, cnt, N);
  finish_mean_k<<<cdiv((long)C * EMBD, 256), 256, 0, stream>>>(semb_o, cnt, C * EMBD);

  // ---- kNN graphs ----
  knn_node_cluster<<<cdiv(N, 128), 128, 1024 * 24 * 4, stream>>>(emb_o, semb_o, bidx, N);
  knn_cluster_cluster<<<cdiv(C, 128), 128, 1024 * 24 * 4, stream>>>(semb_o, sidx);

  // ---- bgraph edges / logits / weights ----
  fill_f32<<<cdiv((long)N, 256), 256, 0, stream>>>(denomB, 0.f, (long)N);
  bgraph_pass1<<<cdiv(nBE, 256), 256, 0, stream>>>(emb_o, semb_o, bidx, cluster,
                                                   bgraph_o, logits_o, bw_o, denomB, nBE, N);
  norm_pass2<<<cdiv(nBE, 256), 256, 0, stream>>>(bgraph_o, bw_o, denomB, nBE);

  // ---- sgraph edges / weights ----
  fill_f32<<<cdiv((long)C, 256), 256, 0, stream>>>(denomS, 0.f, (long)C);
  sgraph_pass1<<<cdiv(nSE, 256), 256, 0, stream>>>(semb_o, sidx, sgraph_o, sw_o, denomS, C * 10);
  norm_pass2<<<cdiv(nSE, 256), 256, 0, stream>>>(sgraph_o, sw_o, denomS, nSE);

  // ---- mask = ones(E) ----
  fill_f32<<<cdiv((long)E, 256), 256, 0, stream>>>(mask_o, 1.f, (long)E);
}
